// MambaVisionBlock_3521873183106
// MI455X (gfx1250) — compile-verified, matrix operand order checked by reading, not run on hardware
//
#include <hip/hip_runtime.h>
#include <hip/hip_bf16.h>

// ---------------------------------------------------------------------------
// MambaVision block for gfx1250 (MI455X).  GEMMs via v_wmma_f32_16x16x32_f16
// (f16 in, f32 accumulate) with GLOBAL_LOAD_ASYNC_TO_LDS staging (ASYNCcnt).
// Scan via VALU + v_exp_f32 with LDS-staged B/C.
// ---------------------------------------------------------------------------

typedef __attribute__((ext_vector_type(16))) _Float16 v16h;
typedef __attribute__((ext_vector_type(8)))  _Float16 v8h;
typedef __attribute__((ext_vector_type(8)))  float    v8f;

#define B_   8
#define C_   384
#define L_   1024          // 32*32
#define DIN  768           // d_inner
#define NST  16            // d_state
#define DTR  24            // dt_rank
#define M_   (B_ * L_)     // 8192 tokens

// ---------------------------------------------------------------------------
// f32 -> f16 convert with zero padding (weights + dt_lo extraction).
// dst is rd x cd row-major; src is rs x cs with leading dim sld.
// ---------------------------------------------------------------------------
__global__ void cvt_pad_f16(const float* __restrict__ src, _Float16* __restrict__ dst,
                            int rd, int cd, int rs, int cs, int sld)
{
    const int idx = blockIdx.x * blockDim.x + threadIdx.x;
    if (idx >= rd * cd) return;
    const int r = idx / cd, c = idx - r * cd;
    float v = (r < rs && c < cs) ? src[(size_t)r * sld + c] : 0.f;
    dst[idx] = (_Float16)v;
}

// ---------------------------------------------------------------------------
// LayerNorm over channels: x [B,C,H,W] -> tok_f16 [B*L, C].
// One wave per token (wave32), lane handles 12 channels, shuffle reduction.
// ---------------------------------------------------------------------------
__global__ __launch_bounds__(256)
void ln_kernel(const float* __restrict__ x, const float* __restrict__ ln_w,
               const float* __restrict__ ln_b, _Float16* __restrict__ tok)
{
    const int lane = threadIdx.x & 31;
    const int wave = threadIdx.x >> 5;
    const int t = blockIdx.x * 8 + wave;          // token 0..8191
    const int b = t >> 10, l = t & 1023;
    const int hh = l >> 5, ww = l & 31;
    const float* xp = x + (((size_t)b * C_) * 32 + hh) * 32 + ww;  // +c*1024

    float v[12];
    float s = 0.f, s2 = 0.f;
#pragma unroll
    for (int i = 0; i < 12; ++i) {
        const int c = i * 32 + lane;
        float f = xp[(size_t)c * (32 * 32)];
        v[i] = f; s += f; s2 += f * f;
    }
#pragma unroll
    for (int o = 16; o > 0; o >>= 1) {
        s  += __shfl_xor(s,  o, 32);
        s2 += __shfl_xor(s2, o, 32);
    }
    const float mean = s * (1.f / C_);
    const float var  = s2 * (1.f / C_) - mean * mean;
    const float rstd = rsqrtf(var + 1e-5f);
#pragma unroll
    for (int i = 0; i < 12; ++i) {
        const int c = i * 32 + lane;
        float xn = (v[i] - mean) * rstd * ln_w[c] + ln_b[c];
        tok[(size_t)t * C_ + c] = (_Float16)xn;
    }
}

// ---------------------------------------------------------------------------
// Tiled WMMA GEMM:  C[M,N] = A[M,K] * B[N,K]^T   (NT, both K-major f16)
// Block = 256 threads (8 waves), block tile 128x64, BK=32.
// Per wave: 32x32 C tile = 4 WMMAs per BK step (A frags reused across N,
// B frags reused across M) -> 2 ds_load_b128 per WMMA.
// Global -> LDS staging via GLOBAL_LOAD_ASYNC_TO_LDS_B128 (ASYNCcnt),
// LDS row stride 40 halves (80B) -> conflict-free 16B fragment loads.
// Assumes M%128==0, K%32==0 (guaranteed by padding); N guarded.
// MODE 0: f32 store   MODE 1: f16 store
// MODE 2: f32 store with residual add + [B,L,C]->[B,C,H,W] transpose
// ---------------------------------------------------------------------------
template<int MODE>
__global__ __launch_bounds__(256)
void gemm_wmma(const _Float16* __restrict__ A, const _Float16* __restrict__ B,
               void* __restrict__ Cout, const float* __restrict__ resid,
               int M, int N, int K, int ldc)
{
    constexpr int BM = 128, BN = 64, BK = 32, LDSS = 40;
    __shared__ _Float16 As[BM * LDSS];   // 10240 B
    __shared__ _Float16 Bs[BN * LDSS];   //  5120 B

    const int tid  = threadIdx.x;
    const int lane = tid & 31;
    const int wave = tid >> 5;
    const int wm = wave & 3;          // wave M sub-tile (32 rows at wm*32)
    const int wn = wave >> 2;         // wave N sub-tile (32 cols at wn*32)
    const int m0 = blockIdx.x * BM;
    const int n0 = blockIdx.y * BN;

    const int lrow = tid >> 2;        // 0..63  cooperative-load row
    const int lcol = (tid & 3) * 8;   // 0/8/16/24 (8 halves = 16B per thread)

    const int h  = lane >> 4;         // lane half-select
    const int ml = lane & 15;

    // LDS byte offsets (generic-pointer low 32 bits == LDS address)
    const unsigned ldsA = (unsigned)(size_t)(void*)As;
    const unsigned ldsB = (unsigned)(size_t)(void*)Bs;

    v8f acc[2][2] = {};

    for (int kt = 0; kt < K; kt += BK) {
        // ---- stage A tile (128 rows x 32 halves): 2 async 16B chunks/thread
#pragma unroll
        for (int hb = 0; hb < 2; ++hb) {
            const int row = lrow + 64 * hb;
            const unsigned dst = ldsA + (unsigned)(row * LDSS + lcol) * 2u;
            const _Float16* src = A + (size_t)(m0 + row) * K + kt + lcol;
            asm volatile("global_load_async_to_lds_b128 %0, %1, off"
                         :: "v"(dst), "v"(src) : "memory");
        }
        // ---- stage B tile (64 rows x 32 halves): 1 chunk/thread, N guarded
        if (n0 + lrow < N) {
            const unsigned dst = ldsB + (unsigned)(lrow * LDSS + lcol) * 2u;
            const _Float16* src = B + (size_t)(n0 + lrow) * K + kt + lcol;
            asm volatile("global_load_async_to_lds_b128 %0, %1, off"
                         :: "v"(dst), "v"(src) : "memory");
        } else {
            v8h z = {};
            *(v8h*)(Bs + lrow * LDSS + lcol) = z;
        }
        // prefetch next K tile of A into cache hierarchy
        if (kt + BK < K)
            __builtin_prefetch(A + (size_t)(m0 + lrow) * K + kt + BK + lcol, 0, 3);

        asm volatile("s_wait_asynccnt 0x0" ::: "memory");
        __syncthreads();

        // ---- fragments per ISA 7.12.2: lane half h holds K=8h..8h+7, 16+8h..
        union F { v16h v; v8h p[2]; };
        F af[2], bf[2];
#pragma unroll
        for (int i = 0; i < 2; ++i) {
            const int ar = (wm * 32 + i * 16 + ml) * LDSS;
            const int br = (wn * 32 + i * 16 + ml) * LDSS;
            af[i].p[0] = *(const v8h*)(As + ar + 8 * h);
            af[i].p[1] = *(const v8h*)(As + ar + 16 + 8 * h);
            bf[i].p[0] = *(const v8h*)(Bs + br + 8 * h);
            bf[i].p[1] = *(const v8h*)(Bs + br + 16 + 8 * h);
        }
#pragma unroll
        for (int i = 0; i < 2; ++i)
#pragma unroll
            for (int j = 0; j < 2; ++j)
                acc[i][j] = __builtin_amdgcn_wmma_f32_16x16x32_f16(
                    false, af[i].v, false, bf[j].v, (short)0, acc[i][j], false, false);
        __syncthreads();
    }

    // ---- store.  C/D layout: lane&15 = N col; VGPR r -> M = r + 8*(lane>>4)
    const int nlane = lane & 15;
#pragma unroll
    for (int i = 0; i < 2; ++i) {
#pragma unroll
        for (int j = 0; j < 2; ++j) {
#pragma unroll
            for (int r = 0; r < 8; ++r) {
                const int mrow = m0 + wm * 32 + i * 16 + 8 * h + r;
                const int nc   = n0 + wn * 32 + j * 16 + nlane;
                if (nc >= N) continue;
                const float v = acc[i][j][r];
                if (MODE == 0) {
                    ((float*)Cout)[(size_t)mrow * ldc + nc] = v;
                } else if (MODE == 1) {
                    ((_Float16*)Cout)[(size_t)mrow * ldc + nc] = (_Float16)v;
                } else {
                    // residual add + [B,L,C] -> [B,C,H,W] transpose (N == C_)
                    const int b = mrow >> 10, l = mrow & 1023;
                    const int hh = l >> 5, ww = l & 31;
                    const size_t idx = (((size_t)b * N + nc) * 32 + hh) * 32 + ww;
                    ((float*)Cout)[idx] = v + resid[idx];
                }
            }
        }
    }
}

// ---------------------------------------------------------------------------
// Causal depthwise conv1d (width 4) + SiLU.  xz [M,1536] f16 (cols 0..767=xin)
// -> u [M,768] f16.  Fully coalesced over d.
// ---------------------------------------------------------------------------
__global__ void conv_silu_kernel(const _Float16* __restrict__ xz,
                                 const float* __restrict__ cw,
                                 const float* __restrict__ cb,
                                 _Float16* __restrict__ u)
{
    const int idx = blockIdx.x * blockDim.x + threadIdx.x;
    if (idx >= M_ * DIN) return;
    const int d = idx % DIN;
    const int m = idx / DIN;
    const int l = m & 1023;
    float acc = cb[d];
#pragma unroll
    for (int j = 0; j < 4; ++j) {
        const int lj = l - 3 + j;
        if (lj >= 0)
            acc += (float)xz[(size_t)(m - 3 + j) * (2 * DIN) + d] * cw[d * 4 + j];
    }
    const float s = acc / (1.f + __expf(-acc));
    u[(size_t)m * DIN + d] = (_Float16)s;
}

// ---------------------------------------------------------------------------
// Selective scan.  Parallel over (b, d): 8*768 = 6144 channels, 24 blocks.
// B/C vectors (shared across all d of a batch) staged in LDS in 32-step chunks.
// Fuses softplus(dt+bias), +u*D skip, silu(z) gate.  yg -> f16 for out_proj.
// ---------------------------------------------------------------------------
__global__ __launch_bounds__(256)
void scan_kernel(const float* __restrict__ dbc,    // [M,64] cols 24..39=B 40..55=C
                 const float* __restrict__ draw,   // [M,768] raw delta (pre-bias)
                 const _Float16* __restrict__ u,   // [M,768]
                 const _Float16* __restrict__ xz,  // [M,1536] (z = cols 768..)
                 const float* __restrict__ A_log,  // [768,16]
                 const float* __restrict__ Dp,     // [768]
                 const float* __restrict__ dtb,    // [768]
                 _Float16* __restrict__ yg)        // [M,768]
{
    constexpr int CH = 32;
    __shared__ float Bs[CH][NST];
    __shared__ float Cs[CH][NST];
    const int tid = threadIdx.x;
    const int b = blockIdx.x / 3;
    const int d = (blockIdx.x % 3) * 256 + tid;

    float a[NST], hst[NST];
#pragma unroll
    for (int n = 0; n < NST; ++n) {
        a[n] = -__expf(A_log[(size_t)d * NST + n]);   // A = -exp(A_log)
        hst[n] = 0.f;
    }
    const float Dd = Dp[d];
    const float bias = dtb[d];
    const int mb = b << 10;

    for (int c0 = 0; c0 < L_; c0 += CH) {
        // cooperatively stage B,C for CH steps (32 floats/step, 1024 total)
#pragma unroll
        for (int q = 0; q < 4; ++q) {
            const int flat = tid + 256 * q;
            const int s = flat >> 5, j = flat & 31;
            const float val = dbc[(size_t)(mb + c0 + s) * 64 + DTR + j];
            if (j < NST) Bs[s][j] = val; else Cs[s][j - NST] = val;
        }
        __syncthreads();

        for (int s = 0; s < CH; ++s) {
            const int m = mb + c0 + s;
            float dt = draw[(size_t)m * DIN + d] + bias;
            dt = (dt > 20.f) ? dt : log1pf(__expf(dt));      // softplus
            const float uv = (float)u[(size_t)m * DIN + d];
            const float du = dt * uv;
            float y = 0.f;
#pragma unroll
            for (int n = 0; n < NST; ++n) {
                hst[n] = hst[n] * __expf(dt * a[n]) + du * Bs[s][n];
                y += hst[n] * Cs[s][n];
            }
            const float z = (float)xz[(size_t)m * (2 * DIN) + DIN + d];
            const float g = z / (1.f + __expf(-z));           // silu(z)
            yg[(size_t)m * DIN + d] = (_Float16)((y + uv * Dd) * g);
        }
        __syncthreads();
    }
}

// ---------------------------------------------------------------------------
extern "C" void kernel_launch(void* const* d_in, const int* in_sizes, int n_in,
                              void* d_out, int out_size, void* d_ws, size_t ws_size,
                              hipStream_t stream)
{
    const float* x      = (const float*)d_in[0];
    const float* ln_w   = (const float*)d_in[1];
    const float* ln_b   = (const float*)d_in[2];
    const float* in_w   = (const float*)d_in[3];   // [1536,384]
    const float* conv_w = (const float*)d_in[4];   // [768,4]
    const float* conv_b = (const float*)d_in[5];
    const float* xp_w   = (const float*)d_in[6];   // [56,768]
    const float* dt_w   = (const float*)d_in[7];   // [768,24]
    const float* dt_b   = (const float*)d_in[8];
    const float* A_log  = (const float*)d_in[9];   // [768,16]
    const float* Dp     = (const float*)d_in[10];
    const float* out_w  = (const float*)d_in[11];  // [384,768]
    float* out = (float*)d_out;

    char* ws = (char*)d_ws;
    size_t o = 0;
    auto alloc = [&](size_t bytes) -> void* {
        void* p = ws + o; o = (o + bytes + 255) & ~(size_t)255; return p;
    };
    _Float16* tok   = (_Float16*)alloc((size_t)M_ * C_ * 2);       // LN tokens
    _Float16* w_in  = (_Float16*)alloc((size_t)1536 * C_ * 2);
    _Float16* xz    = (_Float16*)alloc((size_t)M_ * 2 * DIN * 2);  // in_proj out
    _Float16* uu    = (_Float16*)alloc((size_t)M_ * DIN * 2);      // conv+silu
    _Float16* w_xp  = (_Float16*)alloc((size_t)64 * DIN * 2);      // padded 56->64
    float*    dbc   = (float*)   alloc((size_t)M_ * 64 * 4);       // x_proj out
    _Float16* dtlo  = (_Float16*)alloc((size_t)M_ * 32 * 2);       // padded 24->32
    _Float16* w_dt  = (_Float16*)alloc((size_t)DIN * 32 * 2);      // padded 24->32
    float*    draw  = (float*)   alloc((size_t)M_ * DIN * 4);      // raw delta
    _Float16* yg    = (_Float16*)alloc((size_t)M_ * DIN * 2);      // gated y
    _Float16* w_out = (_Float16*)alloc((size_t)C_ * DIN * 2);

    // 1) weight conversions (K padded to multiples of 32 where needed)
    cvt_pad_f16<<<(1536 * 384 + 255) / 256, 256, 0, stream>>>(in_w, w_in, 1536, 384, 1536, 384, 384);
    cvt_pad_f16<<<(64 * 768 + 255) / 256, 256, 0, stream>>>(xp_w, w_xp, 64, 768, 56, 768, 768);
    cvt_pad_f16<<<(768 * 32 + 255) / 256, 256, 0, stream>>>(dt_w, w_dt, 768, 32, 768, 24, 24);
    cvt_pad_f16<<<(384 * 768 + 255) / 256, 256, 0, stream>>>(out_w, w_out, 384, 768, 384, 768, 768);

    // 2) LayerNorm -> f16 tokens
    ln_kernel<<<M_ / 8, 256, 0, stream>>>(x, ln_w, ln_b, tok);

    // 3) in_proj: [8192,384] x [1536,384]^T -> xz f16 [8192,1536]
    gemm_wmma<1><<<dim3(M_ / 128, 1536 / 64), 256, 0, stream>>>(
        tok, w_in, xz, nullptr, M_, 1536, 384, 1536);

    // 4) causal depthwise conv + SiLU -> u f16
    conv_silu_kernel<<<(M_ * DIN + 255) / 256, 256, 0, stream>>>(xz, conv_w, conv_b, uu);

    // 5) x_proj: [8192,768] x [56,768]^T -> dbc f32 [8192,64] (cols 0..55)
    gemm_wmma<0><<<dim3(M_ / 128, 1), 256, 0, stream>>>(
        uu, w_xp, dbc, nullptr, M_, 56, 768, 64);

    // 6) extract dt_lo (cols 0..23) -> f16 padded to 32
    cvt_pad_f16<<<(M_ * 32 + 255) / 256, 256, 0, stream>>>(dbc, dtlo, M_, 32, M_, 24, 64);

    // 7) dt_proj: [8192,32] x [768,32]^T -> raw delta f32 [8192,768]
    gemm_wmma<0><<<dim3(M_ / 128, DIN / 64), 256, 0, stream>>>(
        dtlo, w_dt, draw, nullptr, M_, DIN, 32, DIN);

    // 8) selective scan + gate -> yg f16
    scan_kernel<<<B_ * 3, 256, 0, stream>>>(dbc, draw, uu, xz, A_log, Dp, dt_b, yg);

    // 9) out_proj + residual + NCHW transpose -> d_out
    gemm_wmma<2><<<dim3(M_ / 128, C_ / 64), 256, 0, stream>>>(
        yg, w_out, out, x, M_, C_, DIN, C_);
}
